// PPI_90305982365768
// MI455X (gfx1250) — matrix operand order repaired and checked by reading, not compiled, not run on hardware
//
#include <hip/hip_runtime.h>
#include <hip/hip_bf16.h>

// ---------------------------------------------------------------------------
// Child-Sum TreeLSTM on MI455X (gfx1250, wave32, WMMA + async-to-LDS).
// Phase 0: zero state. Phase 1: transpose/convert weights -> fp16 [N][K].
// Phase 2: Wx pre-GEMM [32768x256]x[256x1024] with v_wmma_f32_16x16x32_f16.
// Phase 3: persistent single-workgroup scan (512 steps): per-step h_sum tile
//          staged via GLOBAL_LOAD_ASYNC_TO_LDS_B128 from an fp16 mirror,
//          WMMA recurrent GEMMs, fused gates, parent scatter; final head.
// ---------------------------------------------------------------------------

typedef __attribute__((ext_vector_type(16))) _Float16 v16h;
typedef __attribute__((ext_vector_type(8)))  _Float16 v8h;
typedef __attribute__((ext_vector_type(8)))  float    v8f;

#define E_   256
#define H_   256
#define S_   512
#define B_   64
#define C_   2
#define N3H  768
#define NTOT 1024   /* 3H (iou) + H (f) */

// ---- WMMA fragment loaders (wave32 layouts per CDNA5 ISA 7.12.2) ----------
// A 16x32 fp16, row-major source [M][K], stride in elements.
__device__ __forceinline__ v16h frag_A(const _Float16* base, int stride, int lane) {
  int halfo = (lane & 16) ? 8 : 0;
  const _Float16* p = base + (size_t)(lane & 15) * stride + halfo;
  v8h lo = *(const v8h*)(p);
  v8h hi = *(const v8h*)(p + 16);
  return __builtin_shufflevector(lo, hi, 0,1,2,3,4,5,6,7,8,9,10,11,12,13,14,15);
}
// B 32x16 fp16, source stored N-major (BT[N][K]), stride in elements.
__device__ __forceinline__ v16h frag_B(const _Float16* base, int stride, int lane) {
  int halfo = (lane & 16) ? 8 : 0;
  const _Float16* p = base + (size_t)(lane & 15) * stride + halfo;
  v8h lo = *(const v8h*)(p);
  v8h hi = *(const v8h*)(p + 16);
  return __builtin_shufflevector(lo, hi, 0,1,2,3,4,5,6,7,8,9,10,11,12,13,14,15);
}

__device__ __forceinline__ float sig_(float x) { return 1.0f / (1.0f + __expf(-x)); }

// async memory->LDS copy of 16 bytes; tracked by ASYNCcnt
__device__ __forceinline__ void async_copy_b128(unsigned lds_byte_off, const void* gaddr) {
  asm volatile("global_load_async_to_lds_b128 %0, %1, off"
               :: "v"(lds_byte_off), "v"(gaddr) : "memory");
}
__device__ __forceinline__ void wait_asynccnt0() {
  asm volatile("s_wait_asynccnt 0x0" ::: "memory");
}

// ---------------------------------------------------------------------------
__global__ void tl_zero_f32(float* p, long long n) {
  long long i = (long long)blockIdx.x * blockDim.x + threadIdx.x;
  long long st = (long long)gridDim.x * blockDim.x;
  for (; i < n; i += st) p[i] = 0.0f;
}

// WT[n][e] for n<768 = W_iou[e][n], else W_f[e][n-768]; UT likewise from U_iou/U_f.
__global__ void tl_prep_weights(const float* __restrict__ W_iou, const float* __restrict__ W_f,
                                const float* __restrict__ U_iou, const float* __restrict__ U_f,
                                _Float16* __restrict__ WT, _Float16* __restrict__ UT) {
  int idx = blockIdx.x * blockDim.x + threadIdx.x;
  int total = NTOT * E_;
  if (idx < total) {
    int n = idx >> 8, e = idx & 255;
    float v = (n < N3H) ? W_iou[(size_t)e * N3H + n] : W_f[(size_t)e * H_ + (n - N3H)];
    WT[idx] = (_Float16)v;
  } else if (idx < 2 * total) {
    int i2 = idx - total;
    int n = i2 >> 8, h = i2 & 255;
    float v = (n < N3H) ? U_iou[(size_t)h * N3H + n] : U_f[(size_t)h * H_ + (n - N3H)];
    UT[i2] = (_Float16)v;
  }
}

// ---------------------------------------------------------------------------
// Phase 2: Wx = emb(x) @ [W_iou | W_f]  (M=B*S=32768, K=256, N=1024)
__global__ void __launch_bounds__(256)
tl_wx_gemm(const int* __restrict__ x, const float* __restrict__ embed,
           const float* __restrict__ b_iou, const _Float16* __restrict__ WT,
           float* __restrict__ WxIou, float* __restrict__ WxF) {
  __shared__ _Float16 As[64 * E_];      // 32 KB
  const int mblk = blockIdx.x;          // 0..511
  const int nblk = blockIdx.y;          // 0..7
  const int tid = threadIdx.x;

  for (int idx = tid; idx < 64 * E_; idx += 256) {
    int row = idx >> 8, e = idx & 255;
    int m = mblk * 64 + row;
    int b = m >> 9, s = m & (S_ - 1);
    int tok = x[s * B_ + b];                       // x is [S,B]
    As[idx] = (_Float16)embed[(size_t)tok * E_ + e];
  }
  __syncthreads();

  const int wave = tid >> 5, lane = tid & 31;
  const int half = (lane >> 4) & 1, nco = lane & 15;

  for (int t = wave; t < 32; t += 8) {             // 4 tiles / wave
    int mt = t & 3, nt = t >> 2;
    int ng0 = nblk * 128 + nt * 16;
    v8f acc = {};
#pragma unroll
    for (int k0 = 0; k0 < E_; k0 += 32) {
      v16h af = frag_A(As + mt * 16 * E_ + k0, E_, lane);
      v16h bf = frag_B(WT + (size_t)ng0 * E_ + k0, E_, lane);
      acc = __builtin_amdgcn_wmma_f32_16x16x32_f16(false, af, false, bf,
                                                   (short)0, acc, false, false);
    }
    int n = ng0 + nco;
#pragma unroll
    for (int r = 0; r < 8; ++r) {
      int m = mblk * 64 + mt * 16 + r + 8 * half;
      int b = m >> 9, s = m & (S_ - 1);
      float v = acc[r];
      if (n < N3H) WxIou[((size_t)b * S_ + s) * N3H + n] = v + b_iou[n];
      else         WxF[((size_t)b * (S_ + 1) + s) * H_ + (n - N3H)] = v;
    }
  }
}

// ---------------------------------------------------------------------------
// Phase 3: sequential scan. ONE workgroup, 512 threads (16 waves), 512 steps.
__global__ void __launch_bounds__(512)
tl_scan(const int* __restrict__ parents, const float* __restrict__ b_f,
        const _Float16* __restrict__ UT,
        const float* __restrict__ WxIou, const float* __restrict__ WxF,
        float* __restrict__ hsum, const _Float16* __restrict__ hsum16c,
        _Float16* __restrict__ hsum16, float* __restrict__ fc,
        const float* __restrict__ W_out, const float* __restrict__ b_out,
        float* __restrict__ out) {
  __shared__ _Float16 hsA[B_ * H_];     // 32 KB: h_sum[:,j,:] fp16 (WMMA A)
  __shared__ _Float16 hB[B_ * H_];      // 32 KB: h of node j fp16 (WMMA A)
  __shared__ float    cBuf[B_ * H_];    // 64 KB: c of node j fp32

  const int tid = threadIdx.x;
  const int wave = tid >> 5, lane = tid & 31;
  const int half = (lane >> 4) & 1, nco = lane & 15;
  const unsigned hsA_base = (unsigned)(uintptr_t)(void*)hsA;  // LDS byte offset

  for (int j = 0; j < S_; ++j) {
    __threadfence();                    // make prior-step scatter visible
    __syncthreads();

    // stage h_sum[:, j, :] fp16 mirror -> LDS via async copy (ASYNCcnt path).
    // 64 rows x 512B; 2048 x 16B chunks; 4 per thread.
    for (int ch = tid; ch < (B_ * H_ * 2) / 16; ch += 512) {
      int b = ch >> 5;                       // 32 chunks per 512B row
      int off = (ch & 31) * 16;              // byte offset within row
      unsigned lds = hsA_base + (unsigned)(b * (H_ * 2) + off);
      const char* g = (const char*)&hsum16c[((size_t)b * (S_ + 1) + j) * H_] + off;
      async_copy_b128(lds, g);
    }
    wait_asynccnt0();
    __syncthreads();

    // hint next step's Wx rows toward cache
    if (j + 1 < S_) {
      int b = tid & 63;
      __builtin_prefetch(&WxIou[((size_t)b * S_ + (j + 1)) * N3H], 0, 1);
    }

    // iou GEMM: [64x256]x[256x768]; one wave keeps the i/o/u tiles of the
    // same (b,h) region so gates fuse in-register.
    for (int c = wave; c < 64; c += 16) {
      int mt = c >> 4, ht = c & 15;
      v8f ai = {}, ao = {}, au = {};
#pragma unroll
      for (int k0 = 0; k0 < H_; k0 += 32) {
        v16h af = frag_A(hsA + mt * 16 * H_ + k0, H_, lane);
        v16h bi = frag_B(UT + (size_t)(ht * 16) * H_ + k0, H_, lane);
        v16h bo = frag_B(UT + (size_t)(H_ + ht * 16) * H_ + k0, H_, lane);
        v16h bu = frag_B(UT + (size_t)(2 * H_ + ht * 16) * H_ + k0, H_, lane);
        ai = __builtin_amdgcn_wmma_f32_16x16x32_f16(false, af, false, bi, (short)0, ai, false, false);
        ao = __builtin_amdgcn_wmma_f32_16x16x32_f16(false, af, false, bo, (short)0, ao, false, false);
        au = __builtin_amdgcn_wmma_f32_16x16x32_f16(false, af, false, bu, (short)0, au, false, false);
      }
      int h = ht * 16 + nco;
#pragma unroll
      for (int r = 0; r < 8; ++r) {
        int b = mt * 16 + r + 8 * half;
        size_t wbase = ((size_t)b * S_ + j) * N3H;
        float ig = sig_(ai[r] + WxIou[wbase + h]);
        float og = sig_(ao[r] + WxIou[wbase + H_ + h]);
        float ug = tanhf(au[r] + WxIou[wbase + 2 * H_ + h]);
        float cv = ig * ug + fc[((size_t)b * (S_ + 1) + j) * H_ + h];
        float hv = og * tanhf(cv);
        cBuf[b * H_ + h] = cv;
        hB[b * H_ + h] = (_Float16)hv;
      }
    }
    __syncthreads();

    // f GEMM: [64x256]x[256x256] on h, then scatter h and f*c into parent slot
    for (int c = wave; c < 64; c += 16) {
      int mt = c >> 4, ht = c & 15;
      v8f af_acc = {};
#pragma unroll
      for (int k0 = 0; k0 < H_; k0 += 32) {
        v16h af = frag_A(hB + mt * 16 * H_ + k0, H_, lane);
        v16h bf = frag_B(UT + (size_t)(N3H + ht * 16) * H_ + k0, H_, lane);
        af_acc = __builtin_amdgcn_wmma_f32_16x16x32_f16(false, af, false, bf, (short)0, af_acc, false, false);
      }
      int h = ht * 16 + nco;
#pragma unroll
      for (int r = 0; r < 8; ++r) {
        int b = mt * 16 + r + 8 * half;
        int p = parents[b * S_ + j];                 // parents is [B,S]
        float fg = sig_(af_acc[r] + WxF[((size_t)b * (S_ + 1) + p) * H_ + h] + b_f[h]);
        float cv = cBuf[b * H_ + h];
        float hv = (float)hB[b * H_ + h];
        size_t dst = ((size_t)b * (S_ + 1) + p) * H_ + h;
        float ns = hsum[dst] + hv;                   // unique (b,h) writer this step
        hsum[dst]   = ns;
        hsum16[dst] = (_Float16)ns;                  // keep fp16 mirror coherent
        fc[dst]    += fg * cv;
      }
    }
  }

  // final head: sigmoid([c_root | h_root] @ W_out^T + b_out) -> [64,2]
  __syncthreads();
  if (tid < B_ * C_) {
    int b = tid >> 1, cls = tid & 1;
    float acc = b_out[cls];
    for (int k = 0; k < H_; ++k) {
      acc += cBuf[b * H_ + k] * W_out[(size_t)cls * (2 * H_) + k];
      acc += (float)hB[b * H_ + k] * W_out[(size_t)cls * (2 * H_) + H_ + k];
    }
    out[b * C_ + cls] = sig_(acc);
  }
}

// ---------------------------------------------------------------------------
extern "C" void kernel_launch(void* const* d_in, const int* in_sizes, int n_in,
                              void* d_out, int out_size, void* d_ws, size_t ws_size,
                              hipStream_t stream) {
  const int*   x       = (const int*)d_in[0];    // [S,B]
  const int*   parents = (const int*)d_in[1];    // [B,S]
  // d_in[2] = lengths (unused; all trees full length)
  const float* embed   = (const float*)d_in[3];  // [V,E]
  const float* W_iou   = (const float*)d_in[4];  // [E,3H]
  const float* U_iou   = (const float*)d_in[5];  // [H,3H]
  const float* b_iou   = (const float*)d_in[6];  // [3H]
  const float* W_f     = (const float*)d_in[7];  // [E,H]
  const float* U_f     = (const float*)d_in[8];  // [H,H]
  const float* b_f     = (const float*)d_in[9];  // [H]
  const float* W_out   = (const float*)d_in[10]; // [C,2H]
  const float* b_out   = (const float*)d_in[11]; // [C]
  float* out = (float*)d_out;                    // [B,C]

  // workspace layout (256-aligned)
  char* ws = (char*)d_ws;
  size_t off = 0;
  _Float16* WT   = (_Float16*)(ws + off); off += (size_t)NTOT * E_ * 2;            // 512 KB
  _Float16* UT   = (_Float16*)(ws + off); off += (size_t)NTOT * H_ * 2;            // 512 KB
  float* WxIou   = (float*)(ws + off);    off += (size_t)B_ * S_ * N3H * 4;        // 96 MB
  float* WxF     = (float*)(ws + off);    off += (size_t)B_ * (S_ + 1) * H_ * 4;   // 32 MB
  float* hsum    = (float*)(ws + off);    off += (size_t)B_ * (S_ + 1) * H_ * 4;   // 32 MB
  float* fcbuf   = (float*)(ws + off);    off += (size_t)B_ * (S_ + 1) * H_ * 4;   // 32 MB
  _Float16* hs16 = (_Float16*)(ws + off); off += (size_t)B_ * (S_ + 1) * H_ * 2;   // 16 MB
  (void)ws_size; (void)in_sizes; (void)n_in; (void)out_size;

  // Phase 0: zero h_sum + fc (contiguous), fp16 mirror, and WxF (dummy row S)
  long long nstate = 2LL * B_ * (S_ + 1) * H_;
  tl_zero_f32<<<2048, 256, 0, stream>>>(hsum, nstate);
  tl_zero_f32<<<1024, 256, 0, stream>>>((float*)hs16, (long long)B_ * (S_ + 1) * H_ / 2);
  tl_zero_f32<<<1024, 256, 0, stream>>>(WxF, (long long)B_ * (S_ + 1) * H_);

  // Phase 1: weight transpose + fp16 convert
  int wtot = 2 * NTOT * E_;
  tl_prep_weights<<<(wtot + 255) / 256, 256, 0, stream>>>(W_iou, W_f, U_iou, U_f, WT, UT);

  // Phase 2: Wx pre-GEMM
  tl_wx_gemm<<<dim3(S_ * B_ / 64, NTOT / 128), 256, 0, stream>>>(x, embed, b_iou, WT, WxIou, WxF);

  // Phase 3: persistent scan (one workgroup) + head
  tl_scan<<<1, 512, 0, stream>>>(parents, b_f, UT, WxIou, WxF,
                                 hsum, hs16, hs16, fcbuf, W_out, b_out, out);
}
